// MultiHeadCausalAttention_23802708754634
// MI455X (gfx1250) — matrix operand order repaired and checked by reading, not compiled, not run on hardware
//
#include <hip/hip_runtime.h>

#define DEV __device__ __forceinline__

typedef _Float16 v16h __attribute__((ext_vector_type(16)));
typedef _Float16 v8h  __attribute__((ext_vector_type(8)));
typedef float    v8f  __attribute__((ext_vector_type(8)));

constexpr int B  = 4, S = 2048, D = 768, H = 8, Dh = 96;
constexpr int BS = B * S;
constexpr float EPSN = 1e-6f;
constexpr float NEGC = 1e10f;
constexpr float RSQ_DH = 0.10206207261596577f; // 1/sqrt(96)

DEV v16h cat16(v8h lo, v8h hi) {
  return __builtin_shufflevector(lo, hi, 0,1,2,3,4,5,6,7,8,9,10,11,12,13,14,15);
}
DEV v8h ld8(const _Float16* p) { return *(const v8h*)p; }

DEV v8f wmma_f16(v16h a, v16h b, v8f c) {
  return __builtin_amdgcn_wmma_f32_16x16x32_f16(false, a, false, b, (short)0, c,
                                                false, false);
}

// ---------------- fp32 -> fp16 convert of activations -----------------------
__global__ void cvt_x(const float* __restrict__ x, _Float16* __restrict__ xh, int n) {
  int i = blockIdx.x * blockDim.x + threadIdx.x;
  if (i < n) xh[i] = (_Float16)x[i];
}

// ------- transpose W (D x D) into N-major fp16 (Wt[n][k] = W[k][n]) ---------
__global__ void xpose_w(const float* __restrict__ wq, const float* __restrict__ wk,
                        const float* __restrict__ wv, _Float16* __restrict__ wt) {
  int i = blockIdx.x * blockDim.x + threadIdx.x; // 3*D*D
  if (i >= 3 * D * D) return;
  int mat = i / (D * D);
  int r   = i % (D * D);
  int n = r / D, k = r % D;
  const float* w = (mat == 0) ? wq : (mat == 1) ? wk : wv;
  wt[i] = (_Float16)w[(size_t)k * D + n];
}

// ---------------- QKV projection GEMM: one 16x16 tile per wave --------------
// q,k stored [B,H,S,Dh] fp16; v stored transposed [B,H,Dh,S] fp16.
__global__ __launch_bounds__(256)
void qkv_gemm(const _Float16* __restrict__ xh, const _Float16* __restrict__ wt,
              const float* __restrict__ bq, const float* __restrict__ bk,
              const float* __restrict__ bv,
              _Float16* __restrict__ qb, _Float16* __restrict__ kb,
              _Float16* __restrict__ vtb) {
  const int NT = D / 16;   // 48
  const int MT = BS / 16;  // 512
  int wave = blockIdx.x * (blockDim.x >> 5) + (threadIdx.x >> 5);
  int lane = threadIdx.x & 31;
  int g = lane >> 4, n16 = lane & 15;
  int nt  = wave % NT;
  int mt  = (wave / NT) % MT;
  int mat = wave / (NT * MT);  // 0=q 1=k 2=v

  const _Float16* arow = xh + (size_t)(mt * 16 + n16) * D;
  const _Float16* brow = wt + (size_t)mat * D * D + (size_t)(nt * 16 + n16) * D;

  v8f acc = (v8f){};
#pragma unroll 4
  for (int kk = 0; kk < D; kk += 32) {
    // A frag (16x32, row-major source): lanes 0-15 hold K 0..7 / 16..23
    v16h a = cat16(ld8(arow + kk + 8 * g), ld8(arow + kk + 8 * g + 16));
    // B frag (32x16, N-major source): lanes 0-15 hold K 0..7 / 16..23
    v16h bm = cat16(ld8(brow + kk + 8 * g), ld8(brow + kk + 8 * g + 16));
    acc = wmma_f16(a, bm, acc);
  }

  int col = nt * 16 + n16;
  const float* bias = (mat == 0) ? bq : (mat == 1) ? bk : bv;
  float bval = bias[col];
  int h = col / Dh, dh = col % Dh;
#pragma unroll
  for (int r = 0; r < 8; ++r) {
    int m = mt * 16 + r + 8 * g;       // row in [0,BS)
    int bb = m >> 11, s = m & (S - 1); // S = 2048
    float val = acc[r] + bval;
    size_t bh = (size_t)(bb * H + h);
    if (mat == 2) {
      vtb[(bh * Dh + dh) * S + s] = (_Float16)val;
    } else {
      size_t off = (bh * S + s) * Dh + dh;
      if (mat == 0) qb[off] = (_Float16)val;
      else          kb[off] = (_Float16)val;
    }
  }
}

// ---------------- fused causal flash attention ------------------------------
// One wave per (b,h,q-tile of 16 rows); 4 waves/block share a q-tile index so
// causal trip counts match. Key chunks of 32. Output fp32 attn [B,S,D].
__global__ __launch_bounds__(128)
void flash_attn(const _Float16* __restrict__ qb, const _Float16* __restrict__ kb,
                const _Float16* __restrict__ vtb, const float* __restrict__ pad,
                float* __restrict__ attn) {
  __shared__ _Float16 plds[4 * 16 * 32];
  int wave = threadIdx.x >> 5, lane = threadIdx.x & 31;
  int g = lane >> 4, n16 = lane & 15;
  int qt = blockIdx.x;                // 0..127
  int bh = blockIdx.y * 4 + wave;     // 0..31
  int b  = bh >> 3;
  _Float16* myp = plds + wave * (16 * 32);

  // Preload Q fragments (Dh = 96 = 3 x K32)
  const _Float16* qrow = qb + ((size_t)bh * S + qt * 16 + n16) * Dh;
  v16h qf[3];
#pragma unroll
  for (int t = 0; t < 3; ++t)
    qf[t] = cat16(ld8(qrow + 32 * t + 8 * g), ld8(qrow + 32 * t + 8 * g + 16));

  float mrow[8], lrow[8];
  v8f oacc[6];
#pragma unroll
  for (int r = 0; r < 8; ++r) { mrow[r] = -1e30f; lrow[r] = 0.0f; }
#pragma unroll
  for (int t = 0; t < 6; ++t) oacc[t] = (v8f){};

  int nchunks = (qt * 16 + 15) / 32 + 1;
  for (int c = 0; c < nchunks; ++c) {
    int key0 = c * 32 + n16;
    int key1 = key0 + 16;
    const _Float16* k0 = kb + ((size_t)bh * S + key0) * Dh;
    const _Float16* k1 = kb + ((size_t)bh * S + key1) * Dh;

    v8f s0 = (v8f){}, s1 = (v8f){};
#pragma unroll
    for (int t = 0; t < 3; ++t) {
      v16h b0 = cat16(ld8(k0 + 32 * t + 8 * g), ld8(k0 + 32 * t + 8 * g + 16));
      s0 = wmma_f16(qf[t], b0, s0);
      v16h b1 = cat16(ld8(k1 + 32 * t + 8 * g), ld8(k1 + 32 * t + 8 * g + 16));
      s1 = wmma_f16(qf[t], b1, s1);
    }

    float np0 = 1.0f - pad[b * S + key0];  // merge factor from pad mask
    float np1 = 1.0f - pad[b * S + key1];

#pragma unroll
    for (int r = 0; r < 8; ++r) {
      int i = qt * 16 + r + 8 * g;  // global query row
      float m0 = (key0 <= i) ? np0 : 0.0f;
      float m1 = (key1 <= i) ? np1 : 0.0f;
      // reference order: mask BEFORE 1/sqrt(Dh) scaling
      float w0 = (s0[r] * m0 - NEGC * (1.0f - m0)) * RSQ_DH;
      float w1 = (s1[r] * m1 - NEGC * (1.0f - m1)) * RSQ_DH;
      // row max across the 16-lane half-wave that owns this row
      float rm = fmaxf(w0, w1);
#pragma unroll
      for (int off = 8; off >= 1; off >>= 1)
        rm = fmaxf(rm, __shfl_xor(rm, off, 32));
      float mnew  = fmaxf(mrow[r], rm);
      float scale = __expf(mrow[r] - mnew);
      float p0 = __expf(w0 - mnew);
      float p1 = __expf(w1 - mnew);
      float ps = p0 + p1;
#pragma unroll
      for (int off = 8; off >= 1; off >>= 1)
        ps += __shfl_xor(ps, off, 32);
      lrow[r] = lrow[r] * scale + ps;
      mrow[r] = mnew;
#pragma unroll
      for (int t = 0; t < 6; ++t) oacc[t][r] *= scale;
      // stage P tile (C-layout -> row-major LDS) for the PV A-fragment
      myp[(r + 8 * g) * 32 + n16]      = (_Float16)p0;
      myp[(r + 8 * g) * 32 + n16 + 16] = (_Float16)p1;
    }

    // DS ops are in-order per wave; wait for stores, block compiler reordering
    asm volatile("s_wait_dscnt 0" ::: "memory");

    v16h pf = cat16(*(const v8h*)&myp[n16 * 32 + 8 * g],
                    *(const v8h*)&myp[n16 * 32 + 8 * g + 16]);
    const _Float16* vrow =
        vtb + ((size_t)bh * Dh + n16) * S + c * 32 + 8 * g;
#pragma unroll
    for (int t = 0; t < 6; ++t) {
      v16h vb = cat16(ld8(vrow + (size_t)t * 16 * S),
                      ld8(vrow + (size_t)t * 16 * S + 16));
      oacc[t] = wmma_f16(pf, vb, oacc[t]);
    }
    asm volatile("" ::: "memory");
  }

  int h = bh & 7;
#pragma unroll
  for (int t = 0; t < 6; ++t) {
#pragma unroll
    for (int r = 0; r < 8; ++r) {
      int s = qt * 16 + r + 8 * g;
      attn[((size_t)(b * S + s)) * D + h * Dh + t * 16 + n16] =
          oacc[t][r] / lrow[r];
    }
  }
}

// ---------------- residual + LayerNorm --------------------------------------
__global__ __launch_bounds__(256)
void residual_ln(const float* __restrict__ attn, const float* __restrict__ x,
                 const float* __restrict__ gamma, const float* __restrict__ beta,
                 float* __restrict__ out) {
  int row = blockIdx.x;
  const float* ar = attn + (size_t)row * D;
  const float* xr = x + (size_t)row * D;
  float y[3];
  float sum = 0.0f, sq = 0.0f;
#pragma unroll
  for (int kx = 0; kx < 3; ++kx) {
    int col = threadIdx.x + 256 * kx;
    float v = ar[col] + xr[col];
    y[kx] = v;
    sum += v;
    sq += v * v;
  }
#pragma unroll
  for (int off = 16; off >= 1; off >>= 1) {
    sum += __shfl_xor(sum, off, 32);
    sq  += __shfl_xor(sq,  off, 32);
  }
  __shared__ float ssum[8], ssq[8];
  int wave = threadIdx.x >> 5, lane = threadIdx.x & 31;
  if (lane == 0) { ssum[wave] = sum; ssq[wave] = sq; }
  __syncthreads();
  if (wave == 0) {
    float a  = (lane < 8) ? ssum[lane] : 0.0f;
    float q2 = (lane < 8) ? ssq[lane]  : 0.0f;
#pragma unroll
    for (int off = 4; off >= 1; off >>= 1) {
      a  += __shfl_xor(a,  off, 32);
      q2 += __shfl_xor(q2, off, 32);
    }
    if (lane == 0) { ssum[0] = a; ssq[0] = q2; }
  }
  __syncthreads();
  float mu   = ssum[0] / (float)D;
  float var  = ssq[0] / (float)D - mu * mu;
  float rstd = rsqrtf(var + EPSN);
#pragma unroll
  for (int kx = 0; kx < 3; ++kx) {
    int col = threadIdx.x + 256 * kx;
    out[(size_t)row * D + col] = (y[kx] - mu) * rstd * gamma[col] + beta[col];
  }
}

extern "C" void kernel_launch(void* const* d_in, const int* in_sizes, int n_in,
                              void* d_out, int out_size, void* d_ws, size_t ws_size,
                              hipStream_t stream) {
  const float* x     = (const float*)d_in[0];
  const float* pad   = (const float*)d_in[1];
  const float* Wq    = (const float*)d_in[2];
  const float* bq    = (const float*)d_in[3];
  const float* Wk    = (const float*)d_in[4];
  const float* bk    = (const float*)d_in[5];
  const float* Wv    = (const float*)d_in[6];
  const float* bv    = (const float*)d_in[7];
  const float* gamma = (const float*)d_in[8];
  const float* beta  = (const float*)d_in[9];
  float* out = (float*)d_out;

  char* ws = (char*)d_ws;
  size_t off = 0;
  auto alloc = [&](size_t bytes) {
    char* p = ws + off;
    off += (bytes + 255) & ~(size_t)255;
    return p;
  };
  _Float16* xh  = (_Float16*)alloc((size_t)BS * D * 2);      // fp16 x
  _Float16* wt  = (_Float16*)alloc((size_t)3 * D * D * 2);   // fp16 W^T x3
  _Float16* qb  = (_Float16*)alloc((size_t)BS * D * 2);      // Q [B,H,S,Dh]
  _Float16* kb  = (_Float16*)alloc((size_t)BS * D * 2);      // K [B,H,S,Dh]
  _Float16* vtb = (_Float16*)alloc((size_t)BS * D * 2);      // V^T [B,H,Dh,S]
  float*    at  = (float*)   alloc((size_t)BS * D * 4);      // attn out fp32

  cvt_x<<<(BS * D + 255) / 256, 256, 0, stream>>>(x, xh, BS * D);
  xpose_w<<<(3 * D * D + 255) / 256, 256, 0, stream>>>(Wq, Wk, Wv, wt);

  int waves = 3 * (BS / 16) * (D / 16);  // 73728 wave-tiles
  qkv_gemm<<<waves / 8, 256, 0, stream>>>(xh, wt, bq, bk, bv, qb, kb, vtb);

  dim3 ag(S / 16, (B * H) / 4);          // (128 q-tiles, 8 groups of 4 bh)
  flash_attn<<<ag, 128, 0, stream>>>(qb, kb, vtb, pad, at);

  residual_ln<<<BS, 256, 0, stream>>>(at, x, gamma, beta, out);
}
